// MoEFFN_53068615909666
// MI455X (gfx1250) — matrix operand order repaired and checked by reading, not compiled, not run on hardware
//
#include <hip/hip_runtime.h>
#include <stdint.h>
#include <stddef.h>

// ---------------- MoE FFN for MI455X (gfx1250, wave32, WMMA bf16) ----------------
// Routed top-2 MoE: ~283 GFLOP of bf16 WMMA; weights converted once to bf16
// (138 MB, fits the 192 MB L2) so per-tile weight re-reads are L2-resident.
// Wave tile = 32x32 (2 bands x 2 subtiles): A frags reused 4x, B frags 2x.
// A-tile staging uses GLOBAL_LOAD_ASYNC_TO_LDS_B128 (ASYNCcnt) when available.

typedef __attribute__((ext_vector_type(16))) __bf16        v16bf;
typedef __attribute__((ext_vector_type(8)))  float         v8f;
typedef __attribute__((ext_vector_type(8)))  unsigned int  v8u;
typedef __attribute__((ext_vector_type(4)))  int           v4i;

#define T_TOK 8192
#define DIM   1024
#define FFN   2816
#define NEXP  8
#define CAP   8192      // worst-case tokens per expert
#define BM    64
#define BN    128
#define BK    32        // K step; also A-tile LDS stride (16B-aligned rows)
#define BKP   34        // padded LDS k-stride for B (even => dword pair reads)

#if defined(__has_builtin)
#if __has_builtin(__builtin_amdgcn_global_load_async_to_lds_b128)
#define HAVE_ASYNC_LDS 1
#endif
#endif

#ifdef HAVE_ASYNC_LDS
typedef __attribute__((address_space(1))) v4i gv4i_t;   // global int4
typedef __attribute__((address_space(3))) v4i lv4i_t;   // LDS int4
__device__ __forceinline__ void async_ld16(const void* g, void* l) {
    __builtin_amdgcn_global_load_async_to_lds_b128((gv4i_t*)g, (lv4i_t*)l, 0, 0);
}
__device__ __forceinline__ void wait_async0() {
#if __has_builtin(__builtin_amdgcn_s_wait_asynccnt)
    __builtin_amdgcn_s_wait_asynccnt(0);
#else
    asm volatile("s_wait_asynccnt 0x0" ::: "memory");
#endif
}
#endif

__device__ __forceinline__ unsigned short f2bf(float f) {
    union { float f; unsigned int u; } v; v.f = f;
    unsigned int u = v.u;
    if ((u & 0x7fffffffu) > 0x7f800000u) return (unsigned short)((u >> 16) | 0x0040u); // quiet NaN
    return (unsigned short)((u + 0x7fffu + ((u >> 16) & 1u)) >> 16);                   // RNE
}

// ISA 7.12.2: 16-bit A fragment, dword j of lane(half,l16) holds K = ka..ka+1
__device__ __forceinline__ int ka_ofs(int j, int half) {
    return ((j & 4) << 2) + half * 8 + 2 * (j & 3);
}
// 16-bit B fragment: same K order as A (dword j holds K = kb..kb+1)
__device__ __forceinline__ int kb_ofs(int j, int half) {
    return ((j & 4) << 2) + half * 8 + 2 * (j & 3);
}

// ---------------------------------------------------------------------------
__global__ void zero_cnt_kernel(int* cnt) {
    if (threadIdx.x < NEXP) cnt[threadIdx.x] = 0;
}

__global__ void cvt_bf16_kernel(const float4* __restrict__ src,
                                ushort4* __restrict__ dst, int n4) {
    int i = blockIdx.x * blockDim.x + threadIdx.x;
    if (i >= n4) return;
    float4 v = src[i];
    ushort4 r;
    r.x = f2bf(v.x); r.y = f2bf(v.y); r.z = f2bf(v.z); r.w = f2bf(v.w);
    dst[i] = r;
}

// ---------------------------------------------------------------------------
// Router: one wave32 per token. logits = x[t] . Wr[:,e]; top-2; renormalize.
__global__ void router_kernel(const float* __restrict__ x,
                              const float* __restrict__ Wr,
                              float* __restrict__ wts,
                              int* __restrict__ list,
                              int* __restrict__ cnt) {
    int wave = (blockIdx.x * blockDim.x + threadIdx.x) >> 5;
    int lane = threadIdx.x & 31;
    if (wave >= T_TOK) return;
    int t = wave;

    float acc[NEXP];
#pragma unroll
    for (int e = 0; e < NEXP; ++e) acc[e] = 0.0f;

    for (int d = lane; d < DIM; d += 32) {
        float xv = x[(size_t)t * DIM + d];
#pragma unroll
        for (int e = 0; e < NEXP; ++e) acc[e] += xv * Wr[d * NEXP + e];
    }
#pragma unroll
    for (int e = 0; e < NEXP; ++e) {
#pragma unroll
        for (int off = 16; off > 0; off >>= 1)
            acc[e] += __shfl_xor(acc[e], off, 32);
    }
    if (lane == 0) {
        int i0 = 0; float m0 = acc[0];
#pragma unroll
        for (int e = 1; e < NEXP; ++e) if (acc[e] > m0) { m0 = acc[e]; i0 = e; }
        int i1 = -1; float m1 = -3.4e38f;
#pragma unroll
        for (int e = 0; e < NEXP; ++e)
            if (e != i0 && acc[e] > m1) { m1 = acc[e]; i1 = e; }
        // renormalized top-2 softmax: w0 = e^m0/(e^m0+e^m1)
        float w0 = 1.0f / (1.0f + __expf(m1 - m0));
        wts[2 * t + 0] = w0;
        wts[2 * t + 1] = 1.0f - w0;
        int p0 = atomicAdd(&cnt[i0], 1); list[i0 * CAP + p0] = 2 * t + 0;
        int p1 = atomicAdd(&cnt[i1], 1); list[i1 * CAP + p1] = 2 * t + 1;
    }
}

// ---------------------------------------------------------------------------
// Gate+Up: per expert, tile [64 gathered rows] x [128 cols of F].
// 8 waves arranged 2x4: wave = (rowgrp 0..1) x (colgrp 0..3); 32x32 per wave.
__launch_bounds__(256)
__global__ void gateup_kernel(const unsigned short* __restrict__ xb,
                              const unsigned short* __restrict__ wgb,
                              const unsigned short* __restrict__ wub,
                              const int* __restrict__ list,
                              const int* __restrict__ cnt,
                              unsigned short* __restrict__ hbuf) {
    const int e  = blockIdx.z;
    const int n0 = blockIdx.x * BN;
    const int m0 = blockIdx.y * BM;
    const int ce = cnt[e];
    if (m0 >= ce) return;

    __shared__ unsigned short As[BM * BK];        // [row][k], 16B-aligned rows
    __shared__ unsigned short Bg[BN * BKP];       // [col][k] (transposed, padded)
    __shared__ unsigned short Bu[BN * BKP];

    const int tid  = threadIdx.x;
    const int lane = tid & 31, wid = tid >> 5;
    const int rg = wid & 1;                       // row group: 32 rows
    const int cg = wid >> 1;                      // col group: 32 cols
    const int half = lane >> 4, l16 = lane & 15;

    v8f zero;
#pragma unroll
    for (int i = 0; i < 8; ++i) zero[i] = 0.0f;
    v8f accg[2][2], accu[2][2];
#pragma unroll
    for (int b = 0; b < 2; ++b)
#pragma unroll
        for (int s = 0; s < 2; ++s) { accg[b][s] = zero; accu[b][s] = zero; }

    const unsigned short* wg_e = wgb + (size_t)e * DIM * FFN;
    const unsigned short* wu_e = wub + (size_t)e * DIM * FFN;

    // A staging: 64 rows x 4 chunks of 8 bf16 -> one 16B chunk per thread
    const int a_row = tid >> 2;
    const int a_c8  = (tid & 3) * 8;
    const unsigned short* a_src = nullptr;
    {
        int ridx = m0 + a_row;
        if (ridx < ce) {
            int slot = list[e * CAP + ridx];
            a_src = xb + (size_t)(slot >> 1) * DIM + a_c8;
        }
    }
    unsigned short* a_dst = &As[a_row * BK + a_c8];
#ifdef HAVE_ASYNC_LDS
    if (!a_src) {   // rows beyond token count: zero once, never touched again
        unsigned int* z = (unsigned int*)a_dst;
        z[0] = 0u; z[1] = 0u; z[2] = 0u; z[3] = 0u;
    }
#endif

    for (int k0 = 0; k0 < DIM; k0 += BK) {
        // ---- stage A (gathered token rows) ----
#ifdef HAVE_ASYNC_LDS
        if (a_src) async_ld16(a_src + k0, a_dst);
#else
        {
            uint4 v = make_uint4(0u, 0u, 0u, 0u);
            if (a_src) v = *(const uint4*)(a_src + k0);
            *(uint4*)a_dst = v;
        }
#endif
        // ---- stage B transposed: 32 k-rows x 128 cols, 2 chunks/thread ----
#pragma unroll
        for (int it = 0; it < 2; ++it) {
            int cc = tid + it * 256;          // 0..511
            int kr = cc >> 4;                 // 0..31
            int c8 = (cc & 15) * 8;           // 0..120
            uint4 g4 = *(const uint4*)(wg_e + (size_t)(k0 + kr) * FFN + n0 + c8);
            uint4 u4 = *(const uint4*)(wu_e + (size_t)(k0 + kr) * FFN + n0 + c8);
            const unsigned short* gp = (const unsigned short*)&g4;
            const unsigned short* up = (const unsigned short*)&u4;
#pragma unroll
            for (int q = 0; q < 8; ++q) {
                Bg[(c8 + q) * BKP + kr] = gp[q];
                Bu[(c8 + q) * BKP + kr] = up[q];
            }
        }
#ifdef HAVE_ASYNC_LDS
        wait_async0();
#endif
        __syncthreads();

        // ---- A fragments (one per 16-row band) ----
        const unsigned int* As32 = (const unsigned int*)As;
        v16bf af[2];
#pragma unroll
        for (int b = 0; b < 2; ++b) {
            const int arow = rg * 32 + b * 16 + l16;
            v8u araw;
#pragma unroll
            for (int j = 0; j < 8; ++j)
                araw[j] = As32[arow * (BK / 2) + (ka_ofs(j, half) >> 1)];
            af[b] = __builtin_bit_cast(v16bf, araw);
        }

        const unsigned int* Bg32 = (const unsigned int*)Bg;
        const unsigned int* Bu32 = (const unsigned int*)Bu;
#pragma unroll
        for (int s = 0; s < 2; ++s) {
            const int col = cg * 32 + s * 16 + l16;
            v8u graw, uraw;
#pragma unroll
            for (int j = 0; j < 8; ++j) {
                int ko = kb_ofs(j, half);
                graw[j] = Bg32[(col * BKP + ko) >> 1];
                uraw[j] = Bu32[(col * BKP + ko) >> 1];
            }
            v16bf bgf = __builtin_bit_cast(v16bf, graw);
            v16bf buf = __builtin_bit_cast(v16bf, uraw);
#pragma unroll
            for (int b = 0; b < 2; ++b) {     // B fragment reused across bands
                accg[b][s] = __builtin_amdgcn_wmma_f32_16x16x32_bf16(
                    false, af[b], false, bgf, (short)0, accg[b][s], false, false);
                accu[b][s] = __builtin_amdgcn_wmma_f32_16x16x32_bf16(
                    false, af[b], false, buf, (short)0, accu[b][s], false, false);
            }
        }
        __syncthreads();
    }

    // ---- SwiGLU epilogue -> hbuf[slot][col] (bf16) ----
#pragma unroll
    for (int b = 0; b < 2; ++b) {
#pragma unroll
        for (int v = 0; v < 8; ++v) {
            int ridx = m0 + rg * 32 + b * 16 + v + 8 * half;  // VGPR v -> M = v+8*half
            if (ridx >= ce) continue;
            int slot = list[e * CAP + ridx];
            unsigned short* hrow = hbuf + (size_t)slot * FFN;
#pragma unroll
            for (int s = 0; s < 2; ++s) {
                float g = accg[b][s][v], u = accu[b][s][v];
                float h = (g / (1.0f + __expf(-g))) * u;
                hrow[n0 + cg * 32 + s * 16 + l16] = f2bf(h);
            }
        }
    }
}

// ---------------------------------------------------------------------------
// Down projection: obuf[slot][0:1024] = hbuf[slot][:] @ Wd[e]  (f32 out)
__launch_bounds__(256)
__global__ void down_kernel(const unsigned short* __restrict__ hbuf,
                            const unsigned short* __restrict__ wdb,
                            const int* __restrict__ list,
                            const int* __restrict__ cnt,
                            float* __restrict__ obuf) {
    const int e  = blockIdx.z;
    const int n0 = blockIdx.x * BN;
    const int m0 = blockIdx.y * BM;
    const int ce = cnt[e];
    if (m0 >= ce) return;

    __shared__ unsigned short As[BM * BK];
    __shared__ unsigned short Bs[BN * BKP];

    const int tid  = threadIdx.x;
    const int lane = tid & 31, wid = tid >> 5;
    const int rg = wid & 1, cg = wid >> 1;
    const int half = lane >> 4, l16 = lane & 15;

    v8f zero;
#pragma unroll
    for (int i = 0; i < 8; ++i) zero[i] = 0.0f;
    v8f acc[2][2];
#pragma unroll
    for (int b = 0; b < 2; ++b)
#pragma unroll
        for (int s = 0; s < 2; ++s) acc[b][s] = zero;

    const unsigned short* wd_e = wdb + (size_t)e * FFN * DIM;

    const int a_row = tid >> 2;
    const int a_c8  = (tid & 3) * 8;
    const unsigned short* a_src = nullptr;
    {
        int ridx = m0 + a_row;
        if (ridx < ce) {
            int slot = list[e * CAP + ridx];
            a_src = hbuf + (size_t)slot * FFN + a_c8;
        }
    }
    unsigned short* a_dst = &As[a_row * BK + a_c8];
#ifdef HAVE_ASYNC_LDS
    if (!a_src) {
        unsigned int* z = (unsigned int*)a_dst;
        z[0] = 0u; z[1] = 0u; z[2] = 0u; z[3] = 0u;
    }
#endif

    for (int k0 = 0; k0 < FFN; k0 += BK) {
#ifdef HAVE_ASYNC_LDS
        if (a_src) async_ld16(a_src + k0, a_dst);
#else
        {
            uint4 v = make_uint4(0u, 0u, 0u, 0u);
            if (a_src) v = *(const uint4*)(a_src + k0);
            *(uint4*)a_dst = v;
        }
#endif
#pragma unroll
        for (int it = 0; it < 2; ++it) {
            int cc = tid + it * 256;
            int kr = cc >> 4;
            int c8 = (cc & 15) * 8;
            uint4 d4 = *(const uint4*)(wd_e + (size_t)(k0 + kr) * DIM + n0 + c8);
            const unsigned short* dp = (const unsigned short*)&d4;
#pragma unroll
            for (int q = 0; q < 8; ++q) Bs[(c8 + q) * BKP + kr] = dp[q];
        }
#ifdef HAVE_ASYNC_LDS
        wait_async0();
#endif
        __syncthreads();

        const unsigned int* As32 = (const unsigned int*)As;
        v16bf af[2];
#pragma unroll
        for (int b = 0; b < 2; ++b) {
            const int arow = rg * 32 + b * 16 + l16;
            v8u araw;
#pragma unroll
            for (int j = 0; j < 8; ++j)
                araw[j] = As32[arow * (BK / 2) + (ka_ofs(j, half) >> 1)];
            af[b] = __builtin_bit_cast(v16bf, araw);
        }

        const unsigned int* Bs32 = (const unsigned int*)Bs;
#pragma unroll
        for (int s = 0; s < 2; ++s) {
            const int col = cg * 32 + s * 16 + l16;
            v8u braw;
#pragma unroll
            for (int j = 0; j < 8; ++j)
                braw[j] = Bs32[(col * BKP + kb_ofs(j, half)) >> 1];
            v16bf bfrag = __builtin_bit_cast(v16bf, braw);
#pragma unroll
            for (int b = 0; b < 2; ++b)
                acc[b][s] = __builtin_amdgcn_wmma_f32_16x16x32_bf16(
                    false, af[b], false, bfrag, (short)0, acc[b][s], false, false);
        }
        __syncthreads();
    }

#pragma unroll
    for (int b = 0; b < 2; ++b) {
#pragma unroll
        for (int v = 0; v < 8; ++v) {
            int ridx = m0 + rg * 32 + b * 16 + v + 8 * half;
            if (ridx >= ce) continue;
            int slot = list[e * CAP + ridx];
            float* orow = obuf + (size_t)slot * DIM;
#pragma unroll
            for (int s = 0; s < 2; ++s)
                orow[n0 + cg * 32 + s * 16 + l16] = acc[b][s][v];
        }
    }
}

// ---------------------------------------------------------------------------
// out[t] = w0*obuf[2t] + w1*obuf[2t+1]   (writes every output element)
__global__ void combine_kernel(const float4* __restrict__ obuf,
                               const float* __restrict__ wts,
                               float4* __restrict__ out) {
    int idx = blockIdx.x * blockDim.x + threadIdx.x;   // T*D/4 threads
    int t  = idx >> 8;                                 // D/4 = 256 vec4 per token
    int d4 = idx & 255;
    float w0 = wts[2 * t], w1 = wts[2 * t + 1];
    float4 a = obuf[(size_t)(2 * t) * 256 + d4];
    float4 b = obuf[(size_t)(2 * t + 1) * 256 + d4];
    float4 r;
    r.x = w0 * a.x + w1 * b.x;
    r.y = w0 * a.y + w1 * b.y;
    r.z = w0 * a.z + w1 * b.z;
    r.w = w0 * a.w + w1 * b.w;
    out[idx] = r;
}

// ---------------------------------------------------------------------------
extern "C" void kernel_launch(void* const* d_in, const int* in_sizes, int n_in,
                              void* d_out, int out_size, void* d_ws, size_t ws_size,
                              hipStream_t stream) {
    const float* x  = (const float*)d_in[0];   // [4,2048,1024]
    const float* Wr = (const float*)d_in[1];   // [1024,8]
    const float* Wg = (const float*)d_in[2];   // [8,1024,2816]
    const float* Wu = (const float*)d_in[3];   // [8,1024,2816]
    const float* Wd = (const float*)d_in[4];   // [8,2816,1024]
    float* out = (float*)d_out;

    // ---- workspace carve (~315 MB total) ----
    char* p = (char*)d_ws;
    auto carve = [&](size_t bytes) {
        char* r = p;
        p += (bytes + 255) & ~(size_t)255;
        return r;
    };
    const size_t nX  = (size_t)T_TOK * DIM;
    const size_t nW  = (size_t)NEXP * DIM * FFN;
    unsigned short* xb   = (unsigned short*)carve(nX * 2);
    unsigned short* wgb  = (unsigned short*)carve(nW * 2);
    unsigned short* wub  = (unsigned short*)carve(nW * 2);
    unsigned short* wdb  = (unsigned short*)carve(nW * 2);
    unsigned short* hbuf = (unsigned short*)carve((size_t)2 * T_TOK * FFN * 2);
    float*          obuf = (float*)carve((size_t)2 * T_TOK * DIM * 4);
    float*          wts  = (float*)carve((size_t)2 * T_TOK * 4);
    int*            list = (int*)carve((size_t)NEXP * CAP * 4);
    int*            cnt  = (int*)carve(256);

    // 1) reset counters
    zero_cnt_kernel<<<1, 32, 0, stream>>>(cnt);

    // 2) fp32 -> bf16 (x + all expert weights; bf16 weights fit 192 MB L2)
    {
        int n4 = (int)(nX / 4);
        cvt_bf16_kernel<<<(n4 + 255) / 256, 256, 0, stream>>>((const float4*)x, (ushort4*)xb, n4);
        n4 = (int)(nW / 4);
        cvt_bf16_kernel<<<(n4 + 255) / 256, 256, 0, stream>>>((const float4*)Wg, (ushort4*)wgb, n4);
        cvt_bf16_kernel<<<(n4 + 255) / 256, 256, 0, stream>>>((const float4*)Wu, (ushort4*)wub, n4);
        cvt_bf16_kernel<<<(n4 + 255) / 256, 256, 0, stream>>>((const float4*)Wd, (ushort4*)wdb, n4);
    }

    // 3) router (one wave32 per token)
    router_kernel<<<T_TOK / 8, 256, 0, stream>>>(x, Wr, wts, list, cnt);

    // 4) gate+up + SwiGLU  (grid: 22 col-tiles x 128 row-tiles x 8 experts)
    gateup_kernel<<<dim3(FFN / BN, CAP / BM, NEXP), 256, 0, stream>>>(
        xb, wgb, wub, list, cnt, hbuf);

    // 5) down projection  (grid: 8 col-tiles x 128 row-tiles x 8 experts)
    down_kernel<<<dim3(DIM / BN, CAP / BM, NEXP), 256, 0, stream>>>(
        hbuf, wdb, list, cnt, obuf);

    // 6) weighted combine of each token's two expert rows
    combine_kernel<<<(T_TOK * DIM / 4) / 256, 256, 0, stream>>>(
        (const float4*)obuf, wts, (float4*)out);
}
